// SathikNeuralCore_46007689675049
// MI455X (gfx1250) — matrix operand order repaired and checked by reading, not compiled, not run on hardware
//
#include <hip/hip_runtime.h>
#include <hip/hip_bf16.h>

// ---------------------------------------------------------------------------
// Model constants (fixed by the reference)
// ---------------------------------------------------------------------------
#define L_     4
#define D_     512
#define NH     8
#define DK     64          // D_/NH
#define E_     8
#define VOCAB  32000
#define TOPK_  2
#define BATCH  4
#define S_     1024
#define ROWS   (BATCH * S_)   // 4096

typedef __attribute__((ext_vector_type(16))) __bf16 bf16x16;
typedef __attribute__((ext_vector_type(2)))  __bf16 bf16x2;
typedef __attribute__((ext_vector_type(8)))  float  f32x8;

__device__ __forceinline__ f32x8 wmma_bf16(bf16x16 a, bf16x16 b, f32x8 c) {
  // D = A(16x32) * B(32x16) + C, fp32 accumulate
  return __builtin_amdgcn_wmma_f32_16x16x32_bf16(false, a, false, b, (short)0, c,
                                                 false, false);
}

// ---------------------------------------------------------------------------
// GEMM: C[M,N] = act(A[M,K] @ W[K,N] + bias) with optional epilogues.
//   block tile 64x128, 8 waves (2x4), 32x32 per wave = 2x2 WMMA tiles, BK=32.
//   B tile stored TRANSPOSED in LDS ([n][k]) so B-fragment loads are
//   contiguous (2 x ds_load_b128). Staging packs (k,k+1) pairs -> one
//   v_cvt_pk_bf16_f32 + one aligned 4B store per pair.
// ---------------------------------------------------------------------------
#define GF_RELU      1
#define GF_BF16QKV   2   // store bf16 head-major [B,H,S,DK] instead of fp32
#define GF_ROWSCALE  4   // v *= rowScale[row*rsStride]
#define GF_ACCUM     8   // C += v  (else C = v)
#define GF_ADDBATCH  16  // v += addBatch[(row/S)*N + col]

#define GBM 64
#define GBN 128
#define GBK 32
#define APAD 8
#define BPAD 8           // row stride 40 halves = 80B (16B aligned)

__global__ void __launch_bounds__(256)
gemm_kernel(const float* __restrict__ A, const float* __restrict__ W,
            const float* __restrict__ bias, float* __restrict__ C,
            __bf16* __restrict__ Cbf, const float* __restrict__ rowScale,
            int rsStride, const float* __restrict__ addBatch,
            int M, int N, int K, int flags)
{
  __shared__ __bf16 As[GBM][GBK + APAD];   // [m][k]
  __shared__ __bf16 Bs[GBN][GBK + BPAD];   // [n][k]  (transposed!)

  const int tid  = threadIdx.x;
  const int lane = tid & 31;
  const int wid  = tid >> 5;
  const int half = (lane >> 4) & 1;
  const int r    = lane & 15;
  const int wm   = wid >> 2;   // 0..1
  const int wn   = wid & 3;    // 0..3

  const int blockN = blockIdx.x * GBN;
  const int blockM = blockIdx.y * GBM;

  f32x8 acc[2][2];
#pragma unroll
  for (int i = 0; i < 2; ++i)
#pragma unroll
    for (int j = 0; j < 2; ++j)
#pragma unroll
      for (int e = 0; e < 8; ++e) acc[i][j][e] = 0.0f;

  const int aRow = tid >> 2;          // 0..63
  const int aCol = (tid & 3) * 8;     // 0,8,16,24
  const int bK0  = (tid >> 4) * 2;    // 0,2,..,30 (even k pair)
  const int bN0  = (tid & 15) * 8;    // 0..120    (8 n-values)

  for (int kb = 0; kb < K; kb += GBK) {
    // ---- stage A tile (fp32 -> bf16), row-major ----
    const float* ap = A + (size_t)(blockM + aRow) * K + kb + aCol;
#pragma unroll
    for (int i = 0; i < 8; ++i) As[aRow][aCol + i] = (__bf16)ap[i];
    // ---- stage B tile: 2(k) x 8(n) patch per thread; coalesced global
    //      reads, packed 4B transposed LDS stores ----
    const float* wp0 = W + (size_t)(kb + bK0) * N + blockN + bN0;
    const float* wp1 = wp0 + N;
#pragma unroll
    for (int i = 0; i < 8; ++i) {
      bf16x2 pk;
      pk[0] = (__bf16)wp0[i];
      pk[1] = (__bf16)wp1[i];
      *(bf16x2*)&Bs[bN0 + i][bK0] = pk;   // 4B-aligned (bK0 even)
    }

    if (kb + GBK < K) {   // hint next tiles into cache (global_prefetch_b8)
      __builtin_prefetch(ap + GBK, 0, 3);
      __builtin_prefetch(wp0 + (size_t)GBK * N, 0, 3);
    }
    __syncthreads();

    // ---- A fragments: element j -> kk = half*16 + j, row = r (contiguous)
    bf16x16 af[2];
#pragma unroll
    for (int mt = 0; mt < 2; ++mt) {
      const int mrow = wm * 32 + mt * 16 + r;
#pragma unroll
      for (int j = 0; j < 16; ++j) {
        const int kk = half * 16 + j;
        af[mt][j] = As[mrow][kk];
      }
    }
    // ---- B fragments: element t -> kk = half*16 + t, col = r (contiguous)
#pragma unroll
    for (int nt = 0; nt < 2; ++nt) {
      bf16x16 bfm;
      const int ncol = wn * 32 + nt * 16 + r;
#pragma unroll
      for (int t = 0; t < 16; ++t) bfm[t] = Bs[ncol][half * 16 + t];
      acc[0][nt] = wmma_bf16(af[0], bfm, acc[0][nt]);
      acc[1][nt] = wmma_bf16(af[1], bfm, acc[1][nt]);
    }
    __syncthreads();
  }

  // ---- epilogue: C element i -> row = half*8 + i, col = r
#pragma unroll
  for (int mt = 0; mt < 2; ++mt) {
#pragma unroll
    for (int nt = 0; nt < 2; ++nt) {
      const int col = blockN + wn * 32 + nt * 16 + r;
      const float bv = bias ? bias[col] : 0.0f;
#pragma unroll
      for (int i = 0; i < 8; ++i) {
        const int row = blockM + wm * 32 + mt * 16 + half * 8 + i;
        float v = acc[mt][nt][i] + bv;
        if (flags & GF_RELU)     v = fmaxf(v, 0.0f);
        if (flags & GF_ADDBATCH) v += addBatch[(row / S_) * N + col];
        if (flags & GF_ROWSCALE) v *= rowScale[(size_t)row * rsStride];
        if (flags & GF_BF16QKV) {
          const int b = row / S_, s = row % S_, h = col / DK, dd = col % DK;
          Cbf[(((size_t)(b * NH + h)) * S_ + s) * DK + dd] = (__bf16)v;
        } else if (flags & GF_ACCUM) {
          C[(size_t)row * N + col] += v;
        } else {
          C[(size_t)row * N + col] = v;
        }
      }
    }
  }
}

// ---------------------------------------------------------------------------
// Flash attention. Q/K/V bf16 [B*H, S, DK]. Block = (b,h) x 128 q rows.
// Each wave owns 16 q rows; key blocks of 32; online softmax; WMMA everywhere.
// ---------------------------------------------------------------------------
__global__ void __launch_bounds__(256)
attn_kernel(const __bf16* __restrict__ Q, const __bf16* __restrict__ Kd,
            const __bf16* __restrict__ V, float* __restrict__ Aout)
{
  __shared__ __bf16 Ks[32][DK + 8];       // keys x dk
  __shared__ __bf16 Vs[DK][32 + 8];       // dk x keys (transposed)
  __shared__ __bf16 Ps[8][16][32 + 8];    // per-wave P scratch

  const int tid  = threadIdx.x;
  const int lane = tid & 31;
  const int wid  = tid >> 5;
  const int half = (lane >> 4) & 1;
  const int r    = lane & 15;

  const int bh = blockIdx.x;               // 0..B*H-1
  const int b  = bh / NH, h = bh % NH;
  const int qrow0 = blockIdx.y * 128 + wid * 16;

  const __bf16* qbase = Q  + (size_t)bh * S_ * DK;
  const __bf16* kbase = Kd + (size_t)bh * S_ * DK;
  const __bf16* vbase = V  + (size_t)bh * S_ * DK;

  // Q A-fragments for dk=64 (two K=32 steps), resident for the whole pass
  bf16x16 aq[2];
  {
    const __bf16* qp = qbase + (size_t)(qrow0 + r) * DK;
#pragma unroll
    for (int s = 0; s < 2; ++s)
#pragma unroll
      for (int j = 0; j < 16; ++j) {
        const int kk = s * 32 + half * 16 + j;
        aq[s][j] = qp[kk];
      }
  }

  float mrow[8], lrow[8];
  f32x8 out[4];
#pragma unroll
  for (int i = 0; i < 8; ++i) { mrow[i] = -1e30f; lrow[i] = 0.0f; }
#pragma unroll
  for (int nt = 0; nt < 4; ++nt)
#pragma unroll
    for (int i = 0; i < 8; ++i) out[nt][i] = 0.0f;

  const int ldRow = tid >> 3;        // 0..31
  const int ldCol = (tid & 7) * 8;   // 0..56

  for (int kb = 0; kb < S_; kb += 32) {
    { // cooperative K/V staging; V transposed into Vs
      const __bf16* kp = kbase + (size_t)(kb + ldRow) * DK + ldCol;
      const __bf16* vp = vbase + (size_t)(kb + ldRow) * DK + ldCol;
#pragma unroll
      for (int i = 0; i < 8; ++i) Ks[ldRow][ldCol + i] = kp[i];
#pragma unroll
      for (int i = 0; i < 8; ++i) Vs[ldCol + i][ldRow] = vp[i];
    }
    __syncthreads();

    // ---- scores: two 16x16 key tiles, Q(16x64) @ K^T ----
    f32x8 sc[2];
#pragma unroll
    for (int jt = 0; jt < 2; ++jt) {
      f32x8 c;
#pragma unroll
      for (int i = 0; i < 8; ++i) c[i] = 0.0f;
      bf16x16 bk0, bk1;
      const int key = jt * 16 + r;
#pragma unroll
      for (int t = 0; t < 16; ++t) {
        bk0[t] = Ks[key][half * 16 + t];
        bk1[t] = Ks[key][32 + half * 16 + t];
      }
      c = wmma_bf16(aq[0], bk0, c);
      c = wmma_bf16(aq[1], bk1, c);
#pragma unroll
      for (int i = 0; i < 8; ++i) sc[jt][i] = c[i] * 0.125f;  // 1/sqrt(dk)
    }

    // ---- online softmax (row reductions across the 16 lanes of a half) ----
    float pv[2][8];
#pragma unroll
    for (int i = 0; i < 8; ++i) {
      float mx = fmaxf(sc[0][i], sc[1][i]);
#pragma unroll
      for (int off = 8; off >= 1; off >>= 1) mx = fmaxf(mx, __shfl_xor(mx, off, 32));
      const float mnew = fmaxf(mrow[i], mx);
      const float cf = __expf(mrow[i] - mnew);
      const float p0 = __expf(sc[0][i] - mnew);
      const float p1 = __expf(sc[1][i] - mnew);
      float psum = p0 + p1;
#pragma unroll
      for (int off = 8; off >= 1; off >>= 1) psum += __shfl_xor(psum, off, 32);
      lrow[i] = lrow[i] * cf + psum;
      mrow[i] = mnew;
      pv[0][i] = p0; pv[1][i] = p1;
#pragma unroll
      for (int nt = 0; nt < 4; ++nt) out[nt][i] *= cf;
    }

    // ---- re-layout P (C layout -> A fragment layout) via per-wave LDS ----
#pragma unroll
    for (int jt = 0; jt < 2; ++jt)
#pragma unroll
      for (int i = 0; i < 8; ++i)
        Ps[wid][half * 8 + i][jt * 16 + r] = (__bf16)pv[jt][i];

    bf16x16 ap;
#pragma unroll
    for (int j = 0; j < 16; ++j) {
      const int kk = half * 16 + j;
      ap[j] = Ps[wid][r][kk];
    }

    // ---- O += P(16x32) @ V(32x64): 4 dk tiles ----
#pragma unroll
    for (int nt = 0; nt < 4; ++nt) {
      bf16x16 bv;
      const int dcol = nt * 16 + r;
#pragma unroll
      for (int t = 0; t < 16; ++t) bv[t] = Vs[dcol][half * 16 + t];
      out[nt] = wmma_bf16(ap, bv, out[nt]);
    }
    __syncthreads();
  }

  // ---- finalize: divide by l, scatter heads back to [B,S,D] ----
#pragma unroll
  for (int i = 0; i < 8; ++i) {
    const float inv = 1.0f / lrow[i];
    const int srow = qrow0 + half * 8 + i;
    float* op = Aout + ((size_t)(b * S_ + srow)) * D_ + h * DK;
#pragma unroll
    for (int nt = 0; nt < 4; ++nt) op[nt * 16 + r] = out[nt][i] * inv;
  }
}

// ---------------------------------------------------------------------------
// LayerNorm: out = LN(alpha*x + r (+ bvec[b])) * g + beta. Block per row.
// ---------------------------------------------------------------------------
__global__ void __launch_bounds__(256)
ln_kernel(const float* __restrict__ x, const float* __restrict__ rres,
          const float* __restrict__ bvec, const float* __restrict__ g,
          const float* __restrict__ bt, float* __restrict__ out, float alpha)
{
  __shared__ float red[8];
  const int row = blockIdx.x;
  const int b   = row / S_;
  const int tid = threadIdx.x;

  float v[2];
#pragma unroll
  for (int i = 0; i < 2; ++i) {
    const int d = tid + i * 256;
    float t = alpha * x[(size_t)row * D_ + d];
    if (rres) t += rres[(size_t)row * D_ + d];
    if (bvec) t += bvec[b * D_ + d];
    v[i] = t;
  }
  float s = v[0] + v[1];
#pragma unroll
  for (int off = 16; off >= 1; off >>= 1) s += __shfl_xor(s, off, 32);
  if ((tid & 31) == 0) red[tid >> 5] = s;
  __syncthreads();
  float tot = 0.0f;
#pragma unroll
  for (int i = 0; i < 8; ++i) tot += red[i];
  const float mean = tot / D_;
  __syncthreads();
  float vs = (v[0] - mean) * (v[0] - mean) + (v[1] - mean) * (v[1] - mean);
#pragma unroll
  for (int off = 16; off >= 1; off >>= 1) vs += __shfl_xor(vs, off, 32);
  if ((tid & 31) == 0) red[tid >> 5] = vs;
  __syncthreads();
  float vtot = 0.0f;
#pragma unroll
  for (int i = 0; i < 8; ++i) vtot += red[i];
  const float rstd = rsqrtf(vtot / D_ + 1e-5f);
#pragma unroll
  for (int i = 0; i < 2; ++i) {
    const int d = tid + i * 256;
    out[(size_t)row * D_ + d] = (v[i] - mean) * rstd * g[d] + bt[d];
  }
}

// ---------------------------------------------------------------------------
// Embedding gather + batch-indexed positional encoding
// ---------------------------------------------------------------------------
__global__ void embed_pe_kernel(const int* __restrict__ src,
                                const float* __restrict__ embed,
                                float* __restrict__ x)
{
  const int idx = blockIdx.x * blockDim.x + threadIdx.x;   // ROWS*D
  const int row = idx / D_, d = idx % D_;
  const int b = row / S_;
  const int tok = src[row];
  const int p = d >> 1;
  const float div = __expf((float)(2 * p) * (-9.2103403719761836f / (float)D_));
  const float ang = (float)b * div;
  const float pe = (d & 1) ? __cosf(ang) : __sinf(ang);
  x[idx] = embed[(size_t)tok * D_ + d] + pe;
}

// ---------------------------------------------------------------------------
// MoE gate logits (N=8 is too small for WMMA tiles)
// ---------------------------------------------------------------------------
__global__ void gate_kernel(const float* __restrict__ x, const float* __restrict__ gw,
                            const float* __restrict__ gb, float* __restrict__ logits)
{
  const int idx = blockIdx.x * blockDim.x + threadIdx.x;
  if (idx >= ROWS * E_) return;
  const int row = idx >> 3, e = idx & 7;
  float s = gb[e];
  const float* xp = x + (size_t)row * D_;
  for (int d = 0; d < D_; ++d) s += xp[d] * gw[d * E_ + e];
  logits[idx] = s;
}

// exact top-2 + masked softmax (non-selected weights exactly 0)
__global__ void topk_kernel(const float* __restrict__ logits, float* __restrict__ wts)
{
  const int row = blockIdx.x * blockDim.x + threadIdx.x;
  if (row >= ROWS) return;
  float l[E_];
  for (int e = 0; e < E_; ++e) l[e] = logits[row * E_ + e];
  int i1 = 0;
  for (int e = 1; e < E_; ++e) if (l[e] > l[i1]) i1 = e;
  int i2 = -1;
  for (int e = 0; e < E_; ++e) { if (e == i1) continue; if (i2 < 0 || l[e] > l[i2]) i2 = e; }
  const float w2 = __expf(l[i2] - l[i1]);
  const float den = 1.0f + w2;
  for (int e = 0; e < E_; ++e)
    wts[row * E_ + e] = (e == i1) ? (1.0f / den) : ((e == i2) ? (w2 / den) : 0.0f);
}

// mean over sequence: xm[b,d]
__global__ void xmean_kernel(const float* __restrict__ x, float* __restrict__ xm)
{
  const int idx = blockIdx.x * blockDim.x + threadIdx.x;
  if (idx >= BATCH * D_) return;
  const int b = idx / D_, d = idx % D_;
  float s = 0.0f;
  for (int t = 0; t < S_; ++t) s += x[((size_t)b * S_ + t) * D_ + d];
  xm[idx] = s / (float)S_;
}

// tiny emotion MLP: softmax(xm @ w1 + b1) @ w2 + b2
__global__ void emotion_kernel(const float* __restrict__ xm, const float* __restrict__ w1,
                               const float* __restrict__ b1, const float* __restrict__ w2,
                               const float* __restrict__ b2, float* __restrict__ emo)
{
  __shared__ float p[7];
  const int b = blockIdx.x, tid = threadIdx.x;
  if (tid < 7) {
    float s = b1[tid];
    for (int d = 0; d < D_; ++d) s += xm[b * D_ + d] * w1[d * 7 + tid];
    p[tid] = s;
  }
  __syncthreads();
  if (tid == 0) {
    float mx = p[0];
    for (int j = 1; j < 7; ++j) mx = fmaxf(mx, p[j]);
    float e[7], den = 0.0f;
    for (int j = 0; j < 7; ++j) { e[j] = __expf(p[j] - mx); den += e[j]; }
    for (int j = 0; j < 7; ++j) p[j] = e[j] / den;
  }
  __syncthreads();
  for (int d = tid; d < D_; d += blockDim.x) {
    float s = b2[d];
    for (int j = 0; j < 7; ++j) s += p[j] * w2[j * D_ + d];
    emo[b * D_ + d] = s;
  }
}

// memory_vector @ bottom half of mem_w  ->  mp[b,n]
__global__ void memproj_kernel(const float* __restrict__ mv, const float* __restrict__ mw,
                               float* __restrict__ mp)
{
  const int idx = blockIdx.x * blockDim.x + threadIdx.x;
  if (idx >= BATCH * D_) return;
  const int b = idx / D_, n = idx % D_;
  float s = 0.0f;
  for (int d = 0; d < D_; ++d) s += mv[b * D_ + d] * mw[(size_t)(D_ + d) * D_ + n];
  mp[idx] = s;
}

// knowledge filter: xg = x * sigmoid(x @ kw + kb), one wave per row
__global__ void __launch_bounds__(256)
kf_kernel(const float* __restrict__ x, const float* __restrict__ kw,
          const float* __restrict__ kb, float* __restrict__ xg)
{
  const int wave = (blockIdx.x * blockDim.x + threadIdx.x) >> 5;
  const int lane = threadIdx.x & 31;
  if (wave >= ROWS) return;
  const float* xp = x + (size_t)wave * D_;
  float s = 0.0f;
  for (int d = lane; d < D_; d += 32) s += xp[d] * kw[d];
#pragma unroll
  for (int off = 16; off >= 1; off >>= 1) s += __shfl_xor(s, off, 32);
  const float gate = 1.0f / (1.0f + __expf(-(s + kb[0])));
  for (int d = lane; d < D_; d += 32) xg[(size_t)wave * D_ + d] = xp[d] * gate;
}

// ---------------------------------------------------------------------------
// Orchestration
// ---------------------------------------------------------------------------
extern "C" void kernel_launch(void* const* d_in, const int* in_sizes, int n_in,
                              void* d_out, int out_size, void* d_ws, size_t ws_size,
                              hipStream_t stream)
{
  const int*   src    = (const int*)  d_in[0];
  const float* mv     = (const float*)d_in[1];
  const float* embed  = (const float*)d_in[2];
  const float* out_w  = (const float*)d_in[3];
  const float* out_b  = (const float*)d_in[4];
  const float* wq     = (const float*)d_in[5];
  const float* bq     = (const float*)d_in[6];
  const float* wk     = (const float*)d_in[7];
  const float* bk     = (const float*)d_in[8];
  const float* wv     = (const float*)d_in[9];
  const float* bv     = (const float*)d_in[10];
  const float* wo     = (const float*)d_in[11];
  const float* bo     = (const float*)d_in[12];
  const float* gate_w = (const float*)d_in[13];
  const float* gate_b = (const float*)d_in[14];
  const float* ew1    = (const float*)d_in[15];
  const float* eb1    = (const float*)d_in[16];
  const float* ew2    = (const float*)d_in[17];
  const float* eb2    = (const float*)d_in[18];
  const float* em_w1  = (const float*)d_in[19];
  const float* em_b1  = (const float*)d_in[20];
  const float* em_w2  = (const float*)d_in[21];
  const float* em_b2  = (const float*)d_in[22];
  const float* mem_w  = (const float*)d_in[23];
  const float* mem_b  = (const float*)d_in[24];
  const float* kf_w   = (const float*)d_in[25];
  const float* kf_b   = (const float*)d_in[26];
  const float* ff_w1  = (const float*)d_in[27];
  const float* ff_b1  = (const float*)d_in[28];
  const float* ff_w2  = (const float*)d_in[29];
  const float* ff_b2  = (const float*)d_in[30];
  const float* ln1g = (const float*)d_in[31]; const float* ln1b = (const float*)d_in[32];
  const float* ln2g = (const float*)d_in[33]; const float* ln2b = (const float*)d_in[34];
  const float* ln3g = (const float*)d_in[35]; const float* ln3b = (const float*)d_in[36];
  const float* ln4g = (const float*)d_in[37]; const float* ln4b = (const float*)d_in[38];
  const float* ln5g = (const float*)d_in[39]; const float* ln5b = (const float*)d_in[40];

  // workspace layout (~77 MB)
  char* ws = (char*)d_ws;
  float*  x    = (float*)(ws);                                  //  8 MB
  float*  t1   = (float*)(ws + ((size_t)8  << 20));             //  8 MB
  float*  t2   = (float*)(ws + ((size_t)16 << 20));             //  8 MB
  float*  hbuf = (float*)(ws + ((size_t)24 << 20));             // 32 MB
  float*  moe  = (float*)(ws + ((size_t)56 << 20));             //  8 MB
  __bf16* qb   = (__bf16*)(ws + ((size_t)64 << 20));            //  4 MB
  __bf16* kbuf = (__bf16*)(ws + ((size_t)68 << 20));            //  4 MB
  __bf16* vbuf = (__bf16*)(ws + ((size_t)72 << 20));            //  4 MB
  float*  glog = (float*)(ws + ((size_t)76 << 20));
  float*  wts  = glog + (size_t)ROWS * E_;
  float*  xm   = wts  + (size_t)ROWS * E_;
  float*  emo  = xm   + BATCH * D_;
  float*  mp   = emo  + BATCH * D_;

  const dim3 blk(256);
  auto gemm = [&](const float* A, const float* W, const float* bias, float* C,
                  __bf16* Cbf, const float* rs, int rsS, const float* ab,
                  int M, int N, int K, int flags) {
    dim3 grid(N / GBN, M / GBM);
    gemm_kernel<<<grid, blk, 0, stream>>>(A, W, bias, C, Cbf, rs, rsS, ab, M, N, K, flags);
  };
  auto ln = [&](const float* xin, const float* rr, const float* bvec,
                const float* g, const float* bt, float* outp, float alpha) {
    ln_kernel<<<ROWS, blk, 0, stream>>>(xin, rr, bvec, g, bt, outp, alpha);
  };

  embed_pe_kernel<<<(ROWS * D_) / 256, blk, 0, stream>>>(src, embed, x);

  for (int l = 0; l < L_; ++l) {
    const size_t dd = (size_t)D_ * D_;
    // --- self attention ---
    gemm(x, wq + l * dd, bq + l * D_, nullptr, qb,   nullptr, 0, nullptr, ROWS, D_, D_, GF_BF16QKV);
    gemm(x, wk + l * dd, bk + l * D_, nullptr, kbuf, nullptr, 0, nullptr, ROWS, D_, D_, GF_BF16QKV);
    gemm(x, wv + l * dd, bv + l * D_, nullptr, vbuf, nullptr, 0, nullptr, ROWS, D_, D_, GF_BF16QKV);
    attn_kernel<<<dim3(BATCH * NH, S_ / 128), blk, 0, stream>>>(qb, kbuf, vbuf, t1);
    gemm(t1, wo + l * dd, bo + l * D_, t2, nullptr, nullptr, 0, nullptr, ROWS, D_, D_, 0);
    ln(x, t2, nullptr, ln1g + l * D_, ln1b + l * D_, x, 1.0f);

    // --- MoE (dense expert compute, top-2 sparse weighting) ---
    gate_kernel<<<(ROWS * E_) / 256, blk, 0, stream>>>(x, gate_w + (size_t)l * D_ * E_,
                                                       gate_b + l * E_, glog);
    topk_kernel<<<ROWS / 256, blk, 0, stream>>>(glog, wts);
    for (int e = 0; e < E_; ++e) {
      const size_t eo = (size_t)l * E_ + e;
      gemm(x, ew1 + eo * D_ * 2 * D_, eb1 + eo * 2 * D_, hbuf, nullptr,
           nullptr, 0, nullptr, ROWS, 2 * D_, D_, GF_RELU);
      gemm(hbuf, ew2 + eo * 2 * D_ * D_, eb2 + eo * D_, moe, nullptr,
           wts + e, E_, nullptr, ROWS, D_, 2 * D_,
           GF_ROWSCALE | (e ? GF_ACCUM : 0));
    }
    ln(x, moe, nullptr, ln2g + l * D_, ln2b + l * D_, x, 1.0f);

    // --- EmotionNet: LN(2x + emo) ---
    xmean_kernel<<<(BATCH * D_) / 256, blk, 0, stream>>>(x, xm);
    emotion_kernel<<<BATCH, 64, 0, stream>>>(xm, em_w1 + (size_t)l * D_ * 7, em_b1 + l * 7,
                                             em_w2 + (size_t)l * 7 * D_, em_b2 + l * D_, emo);
    ln(x, nullptr, emo, ln3g + l * D_, ln3b + l * D_, x, 2.0f);

    // --- memory fusion: concat(x, mem) @ mw = x@mw_top + mv@mw_bot ---
    memproj_kernel<<<(BATCH * D_) / 256, blk, 0, stream>>>(mv, mem_w + (size_t)l * 2 * dd, mp);
    gemm(x, mem_w + (size_t)l * 2 * dd, mem_b + l * D_, t1, nullptr,
         nullptr, 0, mp, ROWS, D_, D_, GF_ADDBATCH);
    ln(x, t1, nullptr, ln4g + l * D_, ln4b + l * D_, x, 1.0f);

    // --- knowledge filter ---
    kf_kernel<<<(ROWS * 32) / 256, blk, 0, stream>>>(x, kf_w + l * D_, kf_b + l, t1);
    ln(x, t1, nullptr, ln5g + l * D_, ln5b + l * D_, x, 1.0f);

    // --- FFN (reuses norm5 params) ---
    gemm(x, ff_w1 + (size_t)l * D_ * 4 * D_, ff_b1 + l * 4 * D_, hbuf, nullptr,
         nullptr, 0, nullptr, ROWS, 4 * D_, D_, GF_RELU);
    gemm(hbuf, ff_w2 + (size_t)l * 4 * D_ * D_, ff_b2 + l * D_, t1, nullptr,
         nullptr, 0, nullptr, ROWS, D_, 4 * D_, 0);
    ln(x, t1, nullptr, ln5g + l * D_, ln5b + l * D_, x, 1.0f);
  }

  // --- vocab projection ---
  gemm(x, out_w, out_b, (float*)d_out, nullptr, nullptr, 0, nullptr,
       ROWS, VOCAB, D_, 0);
}
